// MyTransformerEncoderLayer_5068061409499
// MI455X (gfx1250) — matrix operand order repaired and checked by reading, not compiled, not run on hardware
//
#include <hip/hip_runtime.h>
#include <hip/hip_bf16.h>

// ---------------------------------------------------------------------------
// MI455X / gfx1250 transformer encoder layer, round 4 (fix async builtin cast).
// Compute bound: ~137 GFLOP vs ~70MB unique traffic (~3us @ 23.3 TB/s HBM).
// All matmuls via v_wmma_f32_16x16x32_bf16 (wave32 WMMA).
// Workgroup-shared operands (GEMM A-tile; attention K/V tiles -- 8x redundant
// per block otherwise) are staged into LDS with GLOBAL_LOAD_ASYNC_TO_LDS_B128
// (ASYNCcnt + s_wait_asynccnt + barrier, double buffered); fragments then
// read via ds_load_b128 from padded LDS (row strides 40/72 -> bank-clean).
// ---------------------------------------------------------------------------

#define D_MODEL 1024
#define NHEAD   16
#define DH      64
#define D_FF    4096
#define SEQ     2048
#define BATCH   2
#define MROWS   (SEQ * BATCH)   // 4096

typedef __attribute__((ext_vector_type(4)))  int    v4i;
typedef __attribute__((ext_vector_type(8)))  float  v8f;
typedef __attribute__((ext_vector_type(16))) __bf16 v16bf;
typedef __attribute__((ext_vector_type(4)))  __bf16 v4bf;

#if defined(__has_builtin)
#if __has_builtin(__builtin_amdgcn_global_load_async_to_lds_b128)
#define HAVE_ASYNC_LDS 1
#else
#define HAVE_ASYNC_LDS 0
#endif
#if __has_builtin(__builtin_amdgcn_s_wait_asynccnt)
#define HAVE_WAIT_ASYNC 1
#else
#define HAVE_WAIT_ASYNC 0
#endif
#else
#define HAVE_ASYNC_LDS 0
#define HAVE_WAIT_ASYNC 0
#endif

union Frag16 { v16bf v; v4i q[2]; };

// Load a 16x32 bf16 A/B fragment: two 16-byte chunks per lane (global or LDS).
__device__ __forceinline__ v16bf ld_frag2(const __bf16* p0, const __bf16* p1) {
  Frag16 u;
  u.q[0] = *(const v4i*)p0;
  u.q[1] = *(const v4i*)p1;
  return u.v;
}

__device__ __forceinline__ v8f wmma_bf16(v16bf a, v16bf b, v8f c) {
  return __builtin_amdgcn_wmma_f32_16x16x32_bf16(
      /*neg_a=*/false, a, /*neg_b=*/false, b,
      /*c_mod=*/(short)0, c, /*reuse_a=*/false, /*reuse_b=*/false);
}

// 16B global -> LDS copy: async (ASYNCcnt) if available, else through VGPRs.
// Builtin takes typed int4 pointers in AS1 (global) / AS3 (LDS).
__device__ __forceinline__ void cp16_g2l(const __bf16* gsrc, __bf16* ldst) {
#if HAVE_ASYNC_LDS
  typedef __attribute__((address_space(1))) v4i* gp_t;
  typedef __attribute__((address_space(3))) v4i* lp_t;
  __builtin_amdgcn_global_load_async_to_lds_b128(
      (gp_t)(v4i*)(void*)const_cast<__bf16*>(gsrc),
      (lp_t)(v4i*)(void*)ldst,
      /*offset=*/0, /*cpol=*/0);
#else
  *(v4i*)ldst = *(const v4i*)gsrc;
#endif
}

template <int N>
__device__ __forceinline__ void wait_async() {
#if HAVE_ASYNC_LDS
#if HAVE_WAIT_ASYNC
  __builtin_amdgcn_s_wait_asynccnt((unsigned short)N);
#else
  asm volatile("s_wait_asynccnt %0" ::"i"(N) : "memory");
#endif
#endif
}

// ds_swizzle cross-lane xor (group-of-32 mode: imm = xor<<10 | and=0x1f)
template <int MASK>
__device__ __forceinline__ float swz_xor(float x) {
  return __int_as_float(
      __builtin_amdgcn_ds_swizzle(__float_as_int(x), (MASK << 10) | 0x1f));
}

// ---------------------------------------------------------------------------
// Conversion kernels
// ---------------------------------------------------------------------------

// x [S,B,D] f32 -> xb [b*S+s, D] bf16 (folds batch-first transpose; coalesced)
__global__ __launch_bounds__(256) void cvt_x_kernel(const float* __restrict__ x,
                                                    __bf16* __restrict__ xb) {
  size_t i = (size_t)blockIdx.x * 256 + threadIdx.x;   // < MROWS*D
  const int m = (int)(i >> 10), d = (int)(i & 1023);
  const int b = m >> 11, s = m & 2047;
  xb[i] = (__bf16)x[((size_t)s * BATCH + b) * D_MODEL + d];
}

// W [K,N] f32 -> Wt [N,K] bf16 via 32x32 LDS tile: coalesced both sides.
__global__ __launch_bounds__(256) void cvt_wT_kernel(const float* __restrict__ W,
                                                     __bf16* __restrict__ Wt,
                                                     int K, int N) {
  __shared__ __bf16 ts[32][36];            // pad 36 -> conflict-free transpose
  const int tid = threadIdx.x;
  const int tiles_n = N >> 5;
  const int tk = blockIdx.x / tiles_n, tn = blockIdx.x % tiles_n;
  const int ty = tid >> 3;                 // 0..31
  const int tx = (tid & 7) * 4;            // 0,4,...,28

  const float4 v = *(const float4*)(W + (size_t)(tk * 32 + ty) * N + tn * 32 + tx);
  ts[ty][tx + 0] = (__bf16)v.x;
  ts[ty][tx + 1] = (__bf16)v.y;
  ts[ty][tx + 2] = (__bf16)v.z;
  ts[ty][tx + 3] = (__bf16)v.w;
  __syncthreads();

  v4bf o;
  o[0] = ts[tx + 0][ty];
  o[1] = ts[tx + 1][ty];
  o[2] = ts[tx + 2][ty];
  o[3] = ts[tx + 3][ty];
  *(v4bf*)(Wt + (size_t)(tn * 32 + ty) * K + tk * 32 + tx) = o;
}

// ---------------------------------------------------------------------------
// WMMA GEMM:  C[M,N] = A[M,K] @ Wt[N,K]^T + bias, optional GELU.
// Block = 8 waves sharing one 64-row A tile (async-staged to LDS, double
// buffered); each wave owns a distinct 64-col B strip loaded from global.
// Per K-step each wave: 4 LDS A-frags x 4 global B-frags -> 16 WMMAs.
// MODE 0: out row-major [M,N] bf16
// MODE 1: out [b,h,s,dh] bf16 (Q/K layout), m=b*S+s, n=h*64+dh
// MODE 2: out [b,h,dh,s] bf16 (V^T layout)
// ---------------------------------------------------------------------------
#define SA_STRIDE 40   // 64-row A tile, 32 K-elems + pad (80B rows, bank-clean)

template <int MODE, bool GELU>
__global__ __launch_bounds__(256) void gemm_bf16_kernel(
    const __bf16* __restrict__ A, const __bf16* __restrict__ Bt,
    const float* __restrict__ bias, __bf16* __restrict__ out,
    int M, int N, int K, float scale) {
  __shared__ __bf16 sA[2][64 * SA_STRIDE];
  const int tid = threadIdx.x;
  const int lane = tid & 31;
  const int L  = lane & 15;     // A: row idx / B,C: column idx
  const int h2 = lane >> 4;     // half selector (K-chunk / M-offset)
  const int wid = tid >> 5;
  const int tilesN = N >> 6, groups = tilesN >> 3;
  const int mt = blockIdx.x / groups;
  const int nt = (blockIdx.x % groups) * 8 + wid;
  const int m0 = mt * 64, n0 = nt * 64;

  // staging map: 256 thr x 16B = one 64x32 bf16 A tile
  const int srow = tid >> 2, schunk = (tid & 3) * 8;
  const __bf16* gA = A + (size_t)(m0 + srow) * K + schunk;
  __bf16* const sdst[2] = {&sA[0][srow * SA_STRIDE + schunk],
                           &sA[1][srow * SA_STRIDE + schunk]};

  const v8f vzero = {0.f, 0.f, 0.f, 0.f, 0.f, 0.f, 0.f, 0.f};
  v8f acc[4][4];
#pragma unroll
  for (int i = 0; i < 4; ++i)
#pragma unroll
    for (int j = 0; j < 4; ++j) acc[i][j] = vzero;

  cp16_g2l(gA, sdst[0]);                       // prologue stage, buffer 0
  for (int kt = 0; kt < K; kt += 32) {
    const int cur = (kt >> 5) & 1;
    const bool more = (kt + 32) < K;
    if (more) cp16_g2l(gA + kt + 32, sdst[cur ^ 1]);
    if (more) wait_async<1>(); else wait_async<0>();
    __syncthreads();                           // A tile [cur] visible to all

    v16bf bfr[4];
#pragma unroll
    for (int t = 0; t < 4; ++t) {
      // B-frag: lane = col n, half h2 picks K range [kt+16h2, +16) (32B)
      const __bf16* brow = Bt + (size_t)(n0 + t * 16 + L) * K + kt + 16 * h2;
      bfr[t] = ld_frag2(brow, brow + 8);
    }
#pragma unroll
    for (int ma = 0; ma < 4; ++ma) {
      // A-frag from LDS: lane = row, half h2 picks K range [16h2, +16)
      const __bf16* ar = &sA[cur][(ma * 16 + L) * SA_STRIDE + 16 * h2];
      v16bf a = ld_frag2(ar, ar + 8);
#pragma unroll
      for (int t = 0; t < 4; ++t) acc[ma][t] = wmma_bf16(a, bfr[t], acc[ma][t]);
    }
    __syncthreads();                           // done reading before re-stage
  }

#pragma unroll
  for (int ma = 0; ma < 4; ++ma)
#pragma unroll
    for (int t = 0; t < 4; ++t) {
      const int n = n0 + t * 16 + L;
      const float bv = bias[n];
#pragma unroll
      for (int r = 0; r < 8; ++r) {
        const int m = m0 + ma * 16 + r + 8 * h2;   // C/D: vgpr r -> M=r(+8)
        float v = acc[ma][t][r] + bv;
        if (GELU) v = 0.5f * v * (1.0f + erff(v * 0.70710678118654752f));
        v *= scale;
        size_t idx;
        if (MODE == 0) {
          idx = (size_t)m * N + n;
        } else {
          const int b = m >> 11, s = m & 2047;
          const int hh = n >> 6, dh = n & 63;
          if (MODE == 1) idx = (((size_t)(b * NHEAD + hh)) * SEQ + s) * DH + dh;
          else           idx = (((size_t)(b * NHEAD + hh)) * DH + dh) * SEQ + s;
        }
        out[idx] = (__bf16)v;
      }
    }
}

// ---------------------------------------------------------------------------
// Flash attention. One wave per (b,h,32-query tile); 8 waves of a block are
// the same head, so the 32-key K tile and V^T tile are async-staged to LDS
// once per block (double buffered) instead of 8x from L2.
// S^T[key,q] = K.Q^T so softmax rows (fixed q) live in one lane's VGPRs ->
// row reduce = in-reg max/sum + one SWAPX16 ds_swizzle.
// ctx^T[dh,q] += V^T . P^T ; 16 WMMAs per 32-key block.
// ---------------------------------------------------------------------------
#define SK_STRIDE 72   // 32 keys x 64 dh (+pad)  -> bank-clean frag reads
#define SV_STRIDE 40   // 64 dh  x 32 keys (+pad)

__global__ __launch_bounds__(256) void attn_kernel(
    const __bf16* __restrict__ Qb,   // [b,h,s,dh]   (pre-scaled 1/sqrt(Dh))
    const __bf16* __restrict__ Kb,   // [b,h,s,dh]
    const __bf16* __restrict__ Vt,   // [b,h,dh,s]
    __bf16* __restrict__ ctx) {      // [b*S+s, h*64+dh] row-major
  __shared__ __bf16 sK[2][32 * SK_STRIDE];
  __shared__ __bf16 sV[2][64 * SV_STRIDE];
  __shared__ __bf16 sctx[8][16 * 72];          // per-wave writeback repack
  const int tid = threadIdx.x;
  const int lane = tid & 31;
  const int L = lane & 15, h2 = lane >> 4;
  const int wid = tid >> 5;
  const int qt = (blockIdx.x & 7) * 8 + wid;   // query tile (of 32) in head
  const int hb = blockIdx.x >> 3;              // b*NHEAD + h (block-uniform)
  const size_t headoff = (size_t)hb * SEQ * DH;
  const __bf16* kbase = Kb + headoff;
  const __bf16* vbase = Vt + headoff;

  // staging maps (per thread, 16B each for K and V tiles)
  const int krow = tid >> 3, kchunk = (tid & 7) * 8;   // 32 x 64
  const int vrow = tid >> 2, vchunk = (tid & 3) * 8;   // 64 x 32
  const __bf16* gK = kbase + (size_t)krow * DH + kchunk;
  const __bf16* gV = vbase + (size_t)vrow * SEQ + vchunk;
  __bf16* const kdst[2] = {&sK[0][krow * SK_STRIDE + kchunk],
                           &sK[1][krow * SK_STRIDE + kchunk]};
  __bf16* const vdst[2] = {&sV[0][vrow * SV_STRIDE + vchunk],
                           &sV[1][vrow * SV_STRIDE + vchunk]};

  // Q^T B-fragments for query subtiles A (q0..15) and B (q16..31), dh 0..63
  const __bf16* qrA = Qb + headoff + (size_t)(qt * 32 + L) * DH + 16 * h2;
  const __bf16* qrB = qrA + 16 * DH;
  const v16bf BqA0 = ld_frag2(qrA, qrA + 8), BqA1 = ld_frag2(qrA + 32, qrA + 40);
  const v16bf BqB0 = ld_frag2(qrB, qrB + 8), BqB1 = ld_frag2(qrB + 32, qrB + 40);

  const v8f vzero = {0.f, 0.f, 0.f, 0.f, 0.f, 0.f, 0.f, 0.f};
  v8f cx[2][4];
#pragma unroll
  for (int qi = 0; qi < 2; ++qi)
#pragma unroll
    for (int t = 0; t < 4; ++t) cx[qi][t] = vzero;
  float mr[2] = {-1e30f, -1e30f}, lr[2] = {0.f, 0.f};

  cp16_g2l(gK, kdst[0]);                       // prologue stage, buffer 0
  cp16_g2l(gV, vdst[0]);
  for (int j = 0; j < SEQ; j += 32) {
    const int cur = (j >> 5) & 1;
    const bool more = (j + 32) < SEQ;
    if (more) {
      cp16_g2l(gK + (size_t)(j + 32) * DH, kdst[cur ^ 1]);
      cp16_g2l(gV + (j + 32), vdst[cur ^ 1]);
    }
    if (more) wait_async<2>(); else wait_async<0>();
    __syncthreads();                           // K/V tiles [cur] visible

    // ---- scores S^T for keys [j,j+32) x queries [0,32): 4 C fragments ----
    v8f X[2][2];                               // [key subtile][q subtile]
#pragma unroll
    for (int kt2 = 0; kt2 < 2; ++kt2) {
      const __bf16* kr = &sK[cur][(kt2 * 16 + L) * SK_STRIDE + 16 * h2];
      const v16bf A0 = ld_frag2(kr, kr + 8);         // dh 0..31
      const v16bf A1 = ld_frag2(kr + 32, kr + 40);   // dh 32..63
      X[kt2][0] = wmma_bf16(A1, BqA1, wmma_bf16(A0, BqA0, vzero));
      X[kt2][1] = wmma_bf16(A1, BqB1, wmma_bf16(A0, BqB0, vzero));
    }

    // ---- online softmax + P^T repack, per query subtile ----
    v16bf Bp[2];
#pragma unroll
    for (int qi = 0; qi < 2; ++qi) {
      float bm = -1e30f;
#pragma unroll
      for (int r = 0; r < 8; ++r) {
        bm = fmaxf(bm, X[0][qi][r]);
        bm = fmaxf(bm, X[1][qi][r]);
      }
      bm = fmaxf(bm, swz_xor<16>(bm));
      const float mnew = fmaxf(mr[qi], bm);
      const float alpha = __expf(mr[qi] - mnew);
      mr[qi] = mnew;
      lr[qi] *= alpha;
#pragma unroll
      for (int t = 0; t < 4; ++t)
#pragma unroll
        for (int r = 0; r < 8; ++r) cx[qi][t][r] *= alpha;

      float e1[8], e2[8];
#pragma unroll
      for (int r = 0; r < 8; ++r) {
        e1[r] = __expf(X[0][qi][r] - mnew);
        e2[r] = __expf(X[1][qi][r] - mnew);
        lr[qi] += e1[r] + e2[r];
      }
      // B vgpr r holds keys {16h2+2r, 16h2+2r+1}; keys 8..15 of each 16-key
      // score fragment live in the opposite lane-half -> SWAPX16 per reg.
#pragma unroll
      for (int r = 0; r < 8; ++r) {
        const float s1 = swz_xor<16>(e1[r]);
        const float s2 = swz_xor<16>(e2[r]);
        const float lo = h2 ? s2 : e1[r];   // keys 16h2 + r
        const float hi = h2 ? e2[r] : s1;   // keys 16h2 + 8 + r
        Bp[qi][r]     = (__bf16)lo;
        Bp[qi][8 + r] = (__bf16)hi;
      }
    }

    // ---- ctx^T += V^T(16dh x 32key) . P^T(32key x 16q), V reused x2 ----
#pragma unroll
    for (int t = 0; t < 4; ++t) {
      const __bf16* vr = &sV[cur][(t * 16 + L) * SV_STRIDE + 16 * h2];
      const v16bf Av = ld_frag2(vr, vr + 8);
      cx[0][t] = wmma_bf16(Av, Bp[0], cx[0][t]);
      cx[1][t] = wmma_bf16(Av, Bp[1], cx[1][t]);
    }
    __syncthreads();                           // done reading before re-stage
  }

  // ---- normalize + writeback (LDS repack -> coalesced b128 stores) ----
  const int b = hb >> 4, h = hb & 15;
  __bf16* sw = &sctx[wid][0];
#pragma unroll
  for (int qi = 0; qi < 2; ++qi) {
    const float ltot = lr[qi] + swz_xor<16>(lr[qi]);
    const float inv = 1.0f / ltot;
#pragma unroll
    for (int t = 0; t < 4; ++t)
#pragma unroll
      for (int r = 0; r < 8; ++r)
        sw[L * 72 + t * 16 + r + 8 * h2] = (__bf16)(cx[qi][t][r] * inv);
    // same-wave LDS ops are ordered -> no barrier needed (wave-private)
    const int q2 = lane >> 1, half = lane & 1;
    const __bf16* srow2 = sw + q2 * 72 + half * 32;
    const v4i d0 = *(const v4i*)(srow2);
    const v4i d1 = *(const v4i*)(srow2 + 8);
    const v4i d2 = *(const v4i*)(srow2 + 16);
    const v4i d3 = *(const v4i*)(srow2 + 24);
    __bf16* crow = ctx + ((size_t)b * SEQ + qt * 32 + qi * 16 + q2) * D_MODEL +
                   h * DH + half * 32;
    *(v4i*)(crow)      = d0;
    *(v4i*)(crow + 8)  = d1;
    *(v4i*)(crow + 16) = d2;
    *(v4i*)(crow + 24) = d3;
  }
}

// ---------------------------------------------------------------------------
// Residual + LayerNorm. Block per row (1024 elems, 256 thr x 4).
// FIRST: resid = x [S,B,D] f32 (transposed read), out -> y f32 + y bf16
// !FIRST: resid = y [M,D] f32, out -> d_out [S,B,D] f32 (transposed write)
// ---------------------------------------------------------------------------
template <bool FIRST>
__global__ __launch_bounds__(256) void ln_kernel(
    const float* __restrict__ resid, const __bf16* __restrict__ add,
    const float* __restrict__ g, const float* __restrict__ be,
    float* __restrict__ out32, __bf16* __restrict__ outbf) {
  const int m = blockIdx.x;
  const int b = m >> 11, s = m & 2047;
  const int tid = threadIdx.x;
  const int lane = tid & 31, wid = tid >> 5;
  const int d = tid * 4;

  const float* rrow = FIRST ? (resid + ((size_t)s * BATCH + b) * D_MODEL)
                            : (resid + (size_t)m * D_MODEL);
  const __bf16* arow = add + (size_t)m * D_MODEL;

  float v[4];
  float sum = 0.f, sq = 0.f;
#pragma unroll
  for (int i = 0; i < 4; ++i) {
    v[i] = rrow[d + i] + (float)arow[d + i];
    sum += v[i];
    sq += v[i] * v[i];
  }
  sum += swz_xor<16>(sum); sq += swz_xor<16>(sq);
  sum += swz_xor<8>(sum);  sq += swz_xor<8>(sq);
  sum += swz_xor<4>(sum);  sq += swz_xor<4>(sq);
  sum += swz_xor<2>(sum);  sq += swz_xor<2>(sq);
  sum += swz_xor<1>(sum);  sq += swz_xor<1>(sq);

  __shared__ float ssum[8], ssq[8];
  if (lane == 0) { ssum[wid] = sum; ssq[wid] = sq; }
  __syncthreads();
  float ts = 0.f, tq = 0.f;
#pragma unroll
  for (int i = 0; i < 8; ++i) { ts += ssum[i]; tq += ssq[i]; }

  const float mu = ts * (1.0f / D_MODEL);
  const float var = tq * (1.0f / D_MODEL) - mu * mu;
  const float rstd = rsqrtf(var + 1e-5f);

  float* orow = FIRST ? (out32 + (size_t)m * D_MODEL)
                      : (out32 + ((size_t)s * BATCH + b) * D_MODEL);
#pragma unroll
  for (int i = 0; i < 4; ++i) {
    const float o = (v[i] - mu) * rstd * g[d + i] + be[d + i];
    orow[d + i] = o;
    if (FIRST) outbf[(size_t)m * D_MODEL + d + i] = (__bf16)o;
  }
}

// ---------------------------------------------------------------------------
// Host launcher
// ---------------------------------------------------------------------------
extern "C" void kernel_launch(void* const* d_in, const int* in_sizes, int n_in,
                              void* d_out, int out_size, void* d_ws, size_t ws_size,
                              hipStream_t stream) {
  const float* x  = (const float*)d_in[0];
  const float* Wq = (const float*)d_in[1];  const float* bq = (const float*)d_in[2];
  const float* Wk = (const float*)d_in[3];  const float* bk = (const float*)d_in[4];
  const float* Wv = (const float*)d_in[5];  const float* bv = (const float*)d_in[6];
  const float* Wo = (const float*)d_in[7];  const float* bo = (const float*)d_in[8];
  const float* W1 = (const float*)d_in[9];  const float* b1 = (const float*)d_in[10];
  const float* W2 = (const float*)d_in[11]; const float* b2 = (const float*)d_in[12];
  const float* g1 = (const float*)d_in[13]; const float* be1 = (const float*)d_in[14];
  const float* g2 = (const float*)d_in[15]; const float* be2 = (const float*)d_in[16];
  float* out = (float*)d_out;

  // Workspace carve-up (all sizes 256B-aligned). ~136 MB total.
  char* ws = (char*)d_ws;
  size_t off = 0;
  auto take = [&](size_t bytes) {
    void* p = ws + off;
    off += (bytes + 255) & ~(size_t)255;
    return p;
  };
  const size_t MD2 = (size_t)MROWS * D_MODEL * 2;      // 8 MB
  __bf16* xb   = (__bf16*)take(MD2);
  __bf16* WqT  = (__bf16*)take((size_t)D_MODEL * D_MODEL * 2);
  __bf16* WkT  = (__bf16*)take((size_t)D_MODEL * D_MODEL * 2);
  __bf16* WvT  = (__bf16*)take((size_t)D_MODEL * D_MODEL * 2);
  __bf16* WoT  = (__bf16*)take((size_t)D_MODEL * D_MODEL * 2);
  __bf16* W1T  = (__bf16*)take((size_t)D_FF * D_MODEL * 2);
  __bf16* W2T  = (__bf16*)take((size_t)D_MODEL * D_FF * 2);
  __bf16* Qb   = (__bf16*)take(MD2);
  __bf16* Kb   = (__bf16*)take(MD2);
  __bf16* Vt   = (__bf16*)take(MD2);
  __bf16* ctx  = (__bf16*)take(MD2);
  __bf16* xatt = (__bf16*)take(MD2);
  float*  y32  = (float*)take((size_t)MROWS * D_MODEL * 4);
  __bf16* ybf  = (__bf16*)take(MD2);
  __bf16* hid  = (__bf16*)take((size_t)MROWS * D_FF * 2);  // 32 MB
  __bf16* z    = (__bf16*)take(MD2);
  (void)ws_size; (void)n_in; (void)in_sizes; (void)out_size;

  // --- stage inputs as bf16 (coalesced, LDS-tiled transposes) ---
  cvt_x_kernel<<<(MROWS * D_MODEL) / 256, 256, 0, stream>>>(x, xb);
  auto tblocks = [](int K, int N) { return (K / 32) * (N / 32); };
  cvt_wT_kernel<<<tblocks(D_MODEL, D_MODEL), 256, 0, stream>>>(Wq, WqT, D_MODEL, D_MODEL);
  cvt_wT_kernel<<<tblocks(D_MODEL, D_MODEL), 256, 0, stream>>>(Wk, WkT, D_MODEL, D_MODEL);
  cvt_wT_kernel<<<tblocks(D_MODEL, D_MODEL), 256, 0, stream>>>(Wv, WvT, D_MODEL, D_MODEL);
  cvt_wT_kernel<<<tblocks(D_MODEL, D_MODEL), 256, 0, stream>>>(Wo, WoT, D_MODEL, D_MODEL);
  cvt_wT_kernel<<<tblocks(D_MODEL, D_FF), 256, 0, stream>>>(W1, W1T, D_MODEL, D_FF);
  cvt_wT_kernel<<<tblocks(D_FF, D_MODEL), 256, 0, stream>>>(W2, W2T, D_FF, D_MODEL);

  // grid: (M/64) row-tiles x (N/64)/8 groups of 8 col-strips, 8 waves/block
  auto gblocks = [](int M, int N) { return (M / 64) * ((N / 64) / 8); };

  // --- QKV projections (Q scaled by 1/sqrt(Dh)) ---
  gemm_bf16_kernel<1, false><<<gblocks(MROWS, D_MODEL), 256, 0, stream>>>(
      xb, WqT, bq, Qb, MROWS, D_MODEL, D_MODEL, 0.125f);
  gemm_bf16_kernel<1, false><<<gblocks(MROWS, D_MODEL), 256, 0, stream>>>(
      xb, WkT, bk, Kb, MROWS, D_MODEL, D_MODEL, 1.0f);
  gemm_bf16_kernel<2, false><<<gblocks(MROWS, D_MODEL), 256, 0, stream>>>(
      xb, WvT, bv, Vt, MROWS, D_MODEL, D_MODEL, 1.0f);

  // --- flash attention: B*H heads x (S/32)/8 blocks ---
  attn_kernel<<<BATCH * NHEAD * (SEQ / 32) / 8, 256, 0, stream>>>(Qb, Kb, Vt, ctx);

  // --- output projection + residual/LN1 ---
  gemm_bf16_kernel<0, false><<<gblocks(MROWS, D_MODEL), 256, 0, stream>>>(
      ctx, WoT, bo, xatt, MROWS, D_MODEL, D_MODEL, 1.0f);
  ln_kernel<true><<<MROWS, 256, 0, stream>>>(x, xatt, g1, be1, y32, ybf);

  // --- FFN ---
  gemm_bf16_kernel<0, true><<<gblocks(MROWS, D_FF), 256, 0, stream>>>(
      ybf, W1T, b1, hid, MROWS, D_FF, D_MODEL, 1.0f);
  gemm_bf16_kernel<0, false><<<gblocks(MROWS, D_MODEL), 256, 0, stream>>>(
      hid, W2T, b2, z, MROWS, D_MODEL, D_FF, 1.0f);
  ln_kernel<false><<<MROWS, 256, 0, stream>>>(y32, z, g2, be2, out, nullptr);
}
